// SchNetRepresentation_78743930404962
// MI455X (gfx1250) — matrix operand order repaired and checked by reading, not compiled, not run on hardware
//
#include <hip/hip_runtime.h>
#include <hip/hip_bf16.h>
#include <math.h>

typedef __attribute__((ext_vector_type(16))) _Float16 v16h;
typedef __attribute__((ext_vector_type(8)))  _Float16 v8h;
typedef __attribute__((ext_vector_type(8)))  float    v8f;

#define N_ATOMS 50000
#define N_PAIRS 1600000
#define NBF 64
#define NRBF 20
#define CUTOFF_F 5.0f
#define LN2_F 0.6931471805599453f

// ---- CDNA5 WMMA 16x16x32 f16 fragment index helpers (CDNA5 ISA) ----
// A 16x32 (MxK), k order used here: lane<16 -> M=lane, K={0..15}; lane>=16 -> M=lane-16, K={16..31}
__device__ __forceinline__ int a_k_of(int i, int lane) {
    return i + ((lane >> 4) << 4);
}
// B 32x16 (KxN), k order used here: k = i + 16*(lane>>4), n = lane&15  -> contiguous in i for row-major tiles
// C/D 16x16 f32: m = i + 8*(lane>>4), n = lane&15

__device__ __forceinline__ v8f wmma_f16(v16h a, v16h b, v8f c) {
    return __builtin_amdgcn_wmma_f32_16x16x32_f16(
        false, a, false, b, (short)0, c, false, false);
}

__device__ __forceinline__ v16h ld_frag(const _Float16* p) {
    return *(const v16h*)p;   // 32B -> 2x ds_load_b128
}

__device__ __forceinline__ float sspf(float x) {
    return fmaxf(x, 0.0f) + log1pf(__expf(-fabsf(x))) - LN2_F;
}

// Stage W^T (W is [64 x 64] row-major f32, W[k][n]) into A-fragment-major f16:
// slot [(ft*2+ks)*512 + lane*16 + i] = W[a_k_of(i,lane)+ks*32][ft*16 + (lane&15)]
__device__ __forceinline__ void stage_wT_frag(const float* __restrict__ W,
                                              _Float16* __restrict__ sWA, int tid) {
    for (int s = tid; s < 4096; s += 256) {
        int tile = s >> 9;            // 0..7
        int ln   = (s >> 4) & 31;
        int i    = s & 15;
        int ft = tile >> 1, ks = tile & 1;
        int k   = a_k_of(i, ln) + ks * 32;
        int col = ft * 16 + (ln & 15);
        sWA[s] = (_Float16)W[k * 64 + col];
    }
}

// Same for Wf1 [20 x 64], K padded to 32 (single k-step): 2048 slots
__device__ __forceinline__ void stage_w1T_frag(const float* __restrict__ Wf1,
                                               _Float16* __restrict__ sWA, int tid) {
    for (int s = tid; s < 2048; s += 256) {
        int ft = s >> 9;              // 0..3
        int ln = (s >> 4) & 31;
        int i  = s & 15;
        int k  = a_k_of(i, ln);
        int col = ft * 16 + (ln & 15);
        sWA[s] = (k < NRBF) ? (_Float16)Wf1[k * 64 + col] : (_Float16)0.0f;
    }
}

// ---------------------------------------------------------------------------
// Kernel 1: per-pair distance + cosine cutoff
// ---------------------------------------------------------------------------
__global__ __launch_bounds__(256) void k_pairs_prep(
    const float* __restrict__ R, const int* __restrict__ idx_i,
    const int* __restrict__ idx_j, float* __restrict__ d_arr,
    float* __restrict__ rcut) {
    int p = blockIdx.x * 256 + threadIdx.x;
    if (p >= N_PAIRS) return;
    int i = idx_i[p], j = idx_j[p];
    float dx = R[i * 3 + 0] - R[j * 3 + 0];
    float dy = R[i * 3 + 1] - R[j * 3 + 1];
    float dz = R[i * 3 + 2] - R[j * 3 + 2];
    float d = sqrtf(dx * dx + dy * dy + dz * dz);
    d_arr[p] = d;
    float rc = 0.5f * (__cosf((float)M_PI * d / CUTOFF_F) + 1.0f);
    rcut[p] = (d < CUTOFF_F) ? rc : 0.0f;
}

// ---------------------------------------------------------------------------
// Kernel 2: embedding lookup x = emb[Z]
// ---------------------------------------------------------------------------
__global__ __launch_bounds__(256) void k_embed(
    const int* __restrict__ Z, const float* __restrict__ emb,
    float* __restrict__ x) {
    int t = blockIdx.x * 256 + threadIdx.x;
    if (t >= N_ATOMS * NBF) return;
    int a = t >> 6, c = t & 63;
    x[t] = emb[Z[a] * NBF + c];
}

// ---------------------------------------------------------------------------
// Kernel 3: h = x @ W + b computed as h^T = W^T(A) x^T(B); 16 rows per wave
// ---------------------------------------------------------------------------
__global__ __launch_bounds__(256) void k_atom_in(
    const float* __restrict__ x, const float* __restrict__ W,
    const float* __restrict__ b, float* __restrict__ h) {
    __shared__ alignas(16) _Float16 sWA[4096];
    __shared__ alignas(16) float    sB[64];
    __shared__ alignas(16) _Float16 sX[8][1024];

    int tid = threadIdx.x;
    stage_wT_frag(W, sWA, tid);
    if (tid < 64) sB[tid] = b[tid];

    int wave = tid >> 5, lane = tid & 31;
    int r = lane & 15, lh = lane >> 4;
    int rowTile = blockIdx.x * 8 + wave;
    bool active = rowTile < (N_ATOMS / 16);
    int rowBase = rowTile * 16;

    if (active) {
        for (int t = lane; t < 1024; t += 32)
            sX[wave][t] = (_Float16)x[rowBase * 64 + t];
    }
    __syncthreads();

    v16h bx0 = ld_frag(&sX[wave][r * 64 + 16 * lh]);
    v16h bx1 = ld_frag(&sX[wave][r * 64 + 32 + 16 * lh]);

    v8f acc[4];
    #pragma unroll
    for (int ft = 0; ft < 4; ++ft) {
        v8f c = {};
        c = wmma_f16(ld_frag(&sWA[(ft * 2 + 0) * 512 + lane * 16]), bx0, c);
        c = wmma_f16(ld_frag(&sWA[(ft * 2 + 1) * 512 + lane * 16]), bx1, c);
        acc[ft] = c;
    }

    if (active) {
        #pragma unroll
        for (int ft = 0; ft < 4; ++ft) {
            int fb = ft * 16 + 8 * lh;
            float bb[8];
            *(float4*)&bb[0] = *(const float4*)&sB[fb];
            *(float4*)&bb[4] = *(const float4*)&sB[fb + 4];
            float4 o0, o1;
            o0.x = acc[ft][0] + bb[0]; o0.y = acc[ft][1] + bb[1];
            o0.z = acc[ft][2] + bb[2]; o0.w = acc[ft][3] + bb[3];
            o1.x = acc[ft][4] + bb[4]; o1.y = acc[ft][5] + bb[5];
            o1.z = acc[ft][6] + bb[6]; o1.w = acc[ft][7] + bb[7];
            float* hp = &h[(rowBase + r) * 64 + fb];
            *(float4*)hp       = o0;
            *(float4*)(hp + 4) = o1;
        }
    }
}

// ---------------------------------------------------------------------------
// Kernel 4: fused pair pipeline (transposed GEMMs):
//   B = f_ij^T (regs, from d) ; A = Wf1^T -> ssp -> A = Wf2^T, B = t^T
//   -> *rcut -> *h[idx_j] -> atomicAdd agg[idx_i]
// One pair row per lane in the epilogue.
// ---------------------------------------------------------------------------
__global__ __launch_bounds__(256) void k_pair(
    const float* __restrict__ d_arr, const float* __restrict__ rcut,
    const int* __restrict__ idx_i, const int* __restrict__ idx_j,
    const float* __restrict__ Wf1, const float* __restrict__ bf1,
    const float* __restrict__ Wf2, const float* __restrict__ bf2,
    const float* __restrict__ h, float* __restrict__ agg) {
    __shared__ alignas(16) _Float16 sW1A[2048];
    __shared__ alignas(16) _Float16 sW2A[4096];
    __shared__ alignas(16) float    sB1[64], sB2[64];
    __shared__ alignas(16) _Float16 sT[8][1024];

    int tid = threadIdx.x;
    stage_w1T_frag(Wf1, sW1A, tid);
    stage_wT_frag(Wf2, sW2A, tid);
    if (tid < 64) { sB1[tid] = bf1[tid]; sB2[tid] = bf2[tid]; }
    __syncthreads();

    int wave = tid >> 5, lane = tid & 31;
    int r = lane & 15, lh = lane >> 4;
    int pairBase = (blockIdx.x * 8 + wave) * 16;
    int p0 = pairBase + r;

    const float delta = CUTOFF_F / (float)(NRBF - 1);
    const float coeff = -0.5f / (delta * delta);
    float dm = d_arr[p0];

    // B fragment = Gaussian RBF features of this lane's pair (K padded 20->32)
    v16h bfij;
    #pragma unroll
    for (int i = 0; i < 16; ++i) {
        int k = i + 16 * lh;
        float t0 = dm - (float)k * delta;
        float v = __expf(coeff * t0 * t0);
        bfij[i] = (k < NRBF) ? (_Float16)v : (_Float16)0.0f;
    }

    // GEMM1 (single K=32 step) + bias + ssp -> row-major t[pair][feature] in LDS
    #pragma unroll
    for (int ft = 0; ft < 4; ++ft) {
        v8f c = {};
        c = wmma_f16(ld_frag(&sW1A[ft * 512 + lane * 16]), bfij, c);
        int fb = ft * 16 + 8 * lh;
        float bb[8];
        *(float4*)&bb[0] = *(const float4*)&sB1[fb];
        *(float4*)&bb[4] = *(const float4*)&sB1[fb + 4];
        v8h o;
        #pragma unroll
        for (int i = 0; i < 8; ++i) o[i] = (_Float16)sspf(c[i] + bb[i]);
        *(v8h*)&sT[wave][r * 64 + fb] = o;   // one ds_store_b128
    }
    __syncthreads();

    // GEMM2: A = Wf2^T fragments, B = t fragments (contiguous v16h loads)
    v16h bt0 = ld_frag(&sT[wave][r * 64 + 16 * lh]);
    v16h bt1 = ld_frag(&sT[wave][r * 64 + 32 + 16 * lh]);
    v8f acc[4];
    #pragma unroll
    for (int ft = 0; ft < 4; ++ft) {
        v8f c = {};
        c = wmma_f16(ld_frag(&sW2A[(ft * 2 + 0) * 512 + lane * 16]), bt0, c);
        c = wmma_f16(ld_frag(&sW2A[(ft * 2 + 1) * 512 + lane * 16]), bt1, c);
        acc[ft] = c;
    }

    // Epilogue: this lane owns features ft*16 + 8*lh + 0..7 of pair p0.
    float rc = rcut[p0];
    if (rc != 0.0f) {
        int jr = idx_j[p0], ir = idx_i[p0];
        #pragma unroll
        for (int ft = 0; ft < 4; ++ft) {
            int fb = ft * 16 + 8 * lh;
            float bb[8], hh[8];
            *(float4*)&bb[0] = *(const float4*)&sB2[fb];
            *(float4*)&bb[4] = *(const float4*)&sB2[fb + 4];
            *(float4*)&hh[0] = *(const float4*)&h[jr * 64 + fb];
            *(float4*)&hh[4] = *(const float4*)&h[jr * 64 + fb + 4];
            #pragma unroll
            for (int i = 0; i < 8; ++i) {
                float msg = (acc[ft][i] + bb[i]) * rc * hh[i];
                __hip_atomic_fetch_add(&agg[ir * 64 + fb + i], msg,
                                       __ATOMIC_RELAXED, __HIP_MEMORY_SCOPE_AGENT);
            }
        }
    }
}

// ---------------------------------------------------------------------------
// Kernel 5: x += ssp(agg @ Wo1 + bo1) @ Wo2 + bo2 (transposed, fused)
// ---------------------------------------------------------------------------
__global__ __launch_bounds__(256) void k_atom_out(
    const float* __restrict__ agg,
    const float* __restrict__ Wo1, const float* __restrict__ bo1,
    const float* __restrict__ Wo2, const float* __restrict__ bo2,
    float* __restrict__ x) {
    __shared__ alignas(16) _Float16 sW1A[4096], sW2A[4096];
    __shared__ alignas(16) float    sB1[64], sB2[64];
    __shared__ alignas(16) _Float16 sX[8][1024];
    __shared__ alignas(16) _Float16 sT[8][1024];

    int tid = threadIdx.x;
    stage_wT_frag(Wo1, sW1A, tid);
    stage_wT_frag(Wo2, sW2A, tid);
    if (tid < 64) { sB1[tid] = bo1[tid]; sB2[tid] = bo2[tid]; }

    int wave = tid >> 5, lane = tid & 31;
    int r = lane & 15, lh = lane >> 4;
    int rowTile = blockIdx.x * 8 + wave;
    bool active = rowTile < (N_ATOMS / 16);
    int rowBase = rowTile * 16;

    if (active) {
        for (int t = lane; t < 1024; t += 32)
            sX[wave][t] = (_Float16)agg[rowBase * 64 + t];
    }
    __syncthreads();

    // GEMM1
    v16h ba0 = ld_frag(&sX[wave][r * 64 + 16 * lh]);
    v16h ba1 = ld_frag(&sX[wave][r * 64 + 32 + 16 * lh]);
    #pragma unroll
    for (int ft = 0; ft < 4; ++ft) {
        v8f c = {};
        c = wmma_f16(ld_frag(&sW1A[(ft * 2 + 0) * 512 + lane * 16]), ba0, c);
        c = wmma_f16(ld_frag(&sW1A[(ft * 2 + 1) * 512 + lane * 16]), ba1, c);
        int fb = ft * 16 + 8 * lh;
        float bb[8];
        *(float4*)&bb[0] = *(const float4*)&sB1[fb];
        *(float4*)&bb[4] = *(const float4*)&sB1[fb + 4];
        v8h o;
        #pragma unroll
        for (int i = 0; i < 8; ++i) o[i] = (_Float16)sspf(c[i] + bb[i]);
        *(v8h*)&sT[wave][r * 64 + fb] = o;
    }
    __syncthreads();

    // GEMM2 + residual
    v16h bt0 = ld_frag(&sT[wave][r * 64 + 16 * lh]);
    v16h bt1 = ld_frag(&sT[wave][r * 64 + 32 + 16 * lh]);
    v8f acc[4];
    #pragma unroll
    for (int ft = 0; ft < 4; ++ft) {
        v8f c = {};
        c = wmma_f16(ld_frag(&sW2A[(ft * 2 + 0) * 512 + lane * 16]), bt0, c);
        c = wmma_f16(ld_frag(&sW2A[(ft * 2 + 1) * 512 + lane * 16]), bt1, c);
        acc[ft] = c;
    }
    if (active) {
        #pragma unroll
        for (int ft = 0; ft < 4; ++ft) {
            int fb = ft * 16 + 8 * lh;
            float bb[8], xx[8];
            *(float4*)&bb[0] = *(const float4*)&sB2[fb];
            *(float4*)&bb[4] = *(const float4*)&sB2[fb + 4];
            float* xp = &x[(rowBase + r) * 64 + fb];
            *(float4*)&xx[0] = *(const float4*)xp;
            *(float4*)&xx[4] = *(const float4*)(xp + 4);
            float4 o0, o1;
            o0.x = xx[0] + acc[ft][0] + bb[0]; o0.y = xx[1] + acc[ft][1] + bb[1];
            o0.z = xx[2] + acc[ft][2] + bb[2]; o0.w = xx[3] + acc[ft][3] + bb[3];
            o1.x = xx[4] + acc[ft][4] + bb[4]; o1.y = xx[5] + acc[ft][5] + bb[5];
            o1.z = xx[6] + acc[ft][6] + bb[6]; o1.w = xx[7] + acc[ft][7] + bb[7];
            *(float4*)xp       = o0;
            *(float4*)(xp + 4) = o1;
        }
    }
}

// ---------------------------------------------------------------------------
extern "C" void kernel_launch(void* const* d_in, const int* in_sizes, int n_in,
                              void* d_out, int out_size, void* d_ws, size_t ws_size,
                              hipStream_t stream) {
    const int*   Z     = (const int*)  d_in[0];
    const float* R     = (const float*)d_in[1];
    const int*   idx_i = (const int*)  d_in[2];
    const int*   idx_j = (const int*)  d_in[3];
    const float* emb   = (const float*)d_in[4];
    const float* W_in  = (const float*)d_in[5];
    const float* b_in  = (const float*)d_in[6];
    const float* Wf1   = (const float*)d_in[7];
    const float* bf1   = (const float*)d_in[8];
    const float* Wf2   = (const float*)d_in[9];
    const float* bf2   = (const float*)d_in[10];
    const float* Wo1   = (const float*)d_in[11];
    const float* bo1   = (const float*)d_in[12];
    const float* Wo2   = (const float*)d_in[13];
    const float* bo2   = (const float*)d_in[14];
    float* x = (float*)d_out;

    // workspace: d[P] | rcut[P] | h[N*64] | agg[N*64]  (~38.4 MB, all f32)
    float* dbuf = (float*)d_ws;
    float* rcut = dbuf + N_PAIRS;
    float* h    = rcut + N_PAIRS;
    float* agg  = h + (size_t)N_ATOMS * NBF;

    k_pairs_prep<<<(N_PAIRS + 255) / 256, 256, 0, stream>>>(R, idx_i, idx_j, dbuf, rcut);
    k_embed<<<(N_ATOMS * NBF + 255) / 256, 256, 0, stream>>>(Z, emb, x);

    const int atomBlocks = (N_ATOMS / 16 + 7) / 8;   // 391
    const int pairBlocks = N_PAIRS / (16 * 8);       // 12500

    for (int l = 0; l < 3; ++l) {
        hipMemsetAsync(agg, 0, (size_t)N_ATOMS * NBF * sizeof(float), stream);
        k_atom_in<<<atomBlocks, 256, 0, stream>>>(
            x, W_in + l * 64 * 64, b_in + l * 64, h);
        k_pair<<<pairBlocks, 256, 0, stream>>>(
            dbuf, rcut, idx_i, idx_j,
            Wf1 + l * NRBF * 64, bf1 + l * 64,
            Wf2 + l * 64 * 64,  bf2 + l * 64, h, agg);
        k_atom_out<<<atomBlocks, 256, 0, stream>>>(
            agg, Wo1 + l * 64 * 64, bo1 + l * 64,
            Wo2 + l * 64 * 64, bo2 + l * 64, x);
    }
}
